// NonLocal_68255620268249
// MI455X (gfx1250) — matrix operand order repaired and checked by reading, not compiled, not run on hardware
//
#include <hip/hip_runtime.h>
#include <hip/hip_bf16.h>
#include <cmath>

typedef __attribute__((ext_vector_type(16))) __bf16 v16bf;
typedef __attribute__((ext_vector_type(8)))  __bf16 v8bf;
typedef __attribute__((ext_vector_type(8)))  float  v8f;
typedef __attribute__((ext_vector_type(4)))  float  v4f;

#define C_CH   64
#define HWDIM  80
#define NPIX   6400          // 80*80
#define KTOT   576           // 9 taps * 64 in-ch   (K = t*64 + ic)
#define NBATCH 2
#define XCOLS  (HWDIM + 2)   // halo column on each side (zero padding)

enum ConvMode { MODE_NC_BF16 = 0, MODE_CN_BF16 = 1, MODE_FINAL_F32 = 2 };

static __device__ __forceinline__ v8f wmma_bf16(v16bf a, v16bf b, v8f c) {
  // D(16x16 f32) = A(16x32 bf16) x B(32x16 bf16) + C
  return __builtin_amdgcn_wmma_f32_16x16x32_bf16(false, a, false, b, (short)0, c,
                                                 false, false);
}

static __device__ __forceinline__ v16bf pack16(v8bf lo, v8bf hi8) {
  v16bf r;
#pragma unroll
  for (int e = 0; e < 8; ++e) { r[e] = lo[e]; r[e + 8] = hi8[e]; }
  return r;
}

// ---------------------------------------------------------------------------
// 3x3 SAME conv as implicit GEMM. Block = 256 thr (8 waves); each wave owns a
// 16x16 (out-ch x pixel) tile. K is remapped to t*64+ic so after full unroll
// each k-step has a compile-time tap (dh,dw) and 16 contiguous channels.
// The LDS input slab carries a zero halo column on each side, so the inner
// loop is branch-free: 4 aligned ds_load_b128 + 1 wmma per k-step.
//   LDS: xs[4][82][64] bf16 input slab (42KB), wl[16][576] bf16 weights (18KB)
// Epilogue: relu(acc + bias), stored per `mode`.
// ---------------------------------------------------------------------------
__global__ __launch_bounds__(256) void conv3x3_wmma(
    const float* __restrict__ x,      // (B, C, N) f32
    const float* __restrict__ w,      // (O=64, I=64, 3, 3) f32
    const float* __restrict__ bias,   // (64,) f32
    const float* __restrict__ resid,  // (B, C, N) f32 or null
    __bf16* __restrict__ out_bf,      // bf16 output (layout per mode) or null
    float* __restrict__ out_f,        // f32 output (mode 2) or null
    int mode)
{
  __shared__ __bf16 xs[4][XCOLS][C_CH];  // [staged row][1+col][ch], halo cols 0
  __shared__ __bf16 wl[16][KTOT];        // [oc-local][K = t*64 + ic]

  const int tid = threadIdx.x;
  const int n0b = blockIdx.x * 128;      // block pixel base
  const int o0  = blockIdx.y * 16;       // out-channel tile base
  const int b   = blockIdx.z;
  const float* xb = x + (size_t)b * C_CH * NPIX;
  const int rowbase = n0b / HWDIM - 1;   // first staged image row (may be -1)

  // ---- stage input slab: 4 rows x (80+halo) cols x 64 ch ----
  {
    const int q  = tid & 3;              // staged row 0..3
    const int ch = tid >> 2;             // channel 0..63
    const int gr = rowbase + q;
    const bool rok = (gr >= 0) && (gr < HWDIM);
    const float* src = xb + (size_t)ch * NPIX + gr * HWDIM;
    xs[q][0][ch]         = (__bf16)0.0f; // left halo
    xs[q][XCOLS - 1][ch] = (__bf16)0.0f; // right halo
#pragma unroll
    for (int j = 0; j < HWDIM; j += 4) {
      v4f v = rok ? *(const v4f*)(src + j) : (v4f){};
      xs[q][j + 1][ch] = (__bf16)v[0];
      xs[q][j + 2][ch] = (__bf16)v[1];
      xs[q][j + 3][ch] = (__bf16)v[2];
      xs[q][j + 4][ch] = (__bf16)v[3];
    }
  }
  // ---- stage weights, remapped K = t*64 + ic (reads w[o][ic*9+t]) ----
  {
    const int oc   = tid >> 4;           // 0..15
    const int part = tid & 15;           // 36 K values each
    const float* wrow = w + (size_t)(o0 + oc) * KTOT;
#pragma unroll
    for (int i = 0; i < KTOT / 16; ++i) {
      const int kidx = part * (KTOT / 16) + i;
      const int t = kidx >> 6, ic = kidx & 63;
      wl[oc][kidx] = (__bf16)wrow[ic * 9 + t];
    }
  }
  __syncthreads();

  const int lane = tid & 31, wid = tid >> 5;
  const int hi = lane >> 4, lx = lane & 15;
  const int np = n0b + wid * 16 + lx;    // this lane's B-matrix column pixel
  const int ph = np / HWDIM, pw = np % HWDIM;
  const int lrow = ph - rowbase;         // staged-row index of the pixel (1..3)

  v8f acc = {};
#pragma unroll
  for (int kk = 0; kk < KTOT; kk += 32) {
    const int t   = kk >> 6;             // tap: compile-time after unroll
    const int dh  = t / 3 - 1;
    const int dw  = t % 3 - 1;
    const int ic0 = (kk & 63) + hi * 16; // first channel of this lane's K run

    // A fragment: lanes 0-15 K kk+{0..15}; lanes 16-31 +16
    const __bf16* wp = &wl[lx][kk + (hi ? 16 : 0)];
    const v16bf afrag = pack16(*(const v8bf*)(wp), *(const v8bf*)(wp + 8));

    // B fragment: 16 consecutive channels at tap pixel (halo makes it safe)
    const __bf16* bp = &xs[lrow + dh][1 + pw + dw][ic0];
    acc = wmma_bf16(afrag, pack16(*(const v8bf*)(bp), *(const v8bf*)(bp + 8)),
                    acc);
  }

  // ---- Epilogue: C/D layout row = r + 8*hi, col = lx ----
#pragma unroll
  for (int r = 0; r < 8; ++r) {
    const int oc = o0 + r + hi * 8;
    float v = fmaxf(acc[r] + bias[oc], 0.0f);
    if (mode == MODE_NC_BF16) {         // theta/phi: (B, N, C) bf16
      out_bf[((size_t)b * NPIX + np) * C_CH + oc] = (__bf16)v;
    } else if (mode == MODE_CN_BF16) {  // g: (B, C, N) bf16
      out_bf[((size_t)b * C_CH + oc) * NPIX + np] = (__bf16)v;
    } else {                            // final: f32 + residual -> d_out
      const size_t idx = ((size_t)b * C_CH + oc) * NPIX + np;
      out_f[idx] = v + resid[idx];
    }
  }
}

// ---------------------------------------------------------------------------
// Fused non-local attention (flash-style, the 6400x6400 f never hits HBM).
// One wave owns 16 query rows n0..n0+15; loops m over N in steps of 32:
//   s(16x32)  = theta_T(16x64) x phi(64x32)        (4 wmma)
//   online softmax in f32 registers (row reduce via shfl_xor over 16 lanes)
//   out(16x64) += p(16x32) x g(32x64)              (4 wmma)
// p is re-laid-out C/D -> A fragment through a per-wave LDS tile.
// ---------------------------------------------------------------------------
__global__ __launch_bounds__(128) void nonlocal_attn_wmma(
    const __bf16* __restrict__ thetaT,  // (B, N, C) bf16
    const __bf16* __restrict__ phiT,    // (B, N, C) bf16
    const __bf16* __restrict__ gCN,     // (B, C, N) bf16
    float* __restrict__ yCN)            // (B, C, N) f32
{
  __shared__ __bf16 pstage[4][16][32];  // per-wave probability staging

  const int lane = threadIdx.x & 31;
  const int wid  = threadIdx.x >> 5;
  const int hi   = lane >> 4;
  const int lx   = lane & 15;

  const int tile = blockIdx.x * 4 + wid;       // 800 tiles = 2 * 6400/16
  const int b    = tile / (NPIX / 16);
  const int n0   = (tile % (NPIX / 16)) * 16;

  const __bf16* th = thetaT + (size_t)b * NPIX * C_CH;
  const __bf16* pp = phiT   + (size_t)b * NPIX * C_CH;
  const __bf16* gg = gCN    + (size_t)b * C_CH * NPIX;

  // Theta A fragments for the whole row-tile (K = c, two 32-chunks)
  const __bf16* trow = th + (size_t)(n0 + lx) * C_CH + (hi ? 16 : 0);
  const v16bf a0 = pack16(*(const v8bf*)(trow + 0),  *(const v8bf*)(trow + 8));
  const v16bf a1 = pack16(*(const v8bf*)(trow + 32), *(const v8bf*)(trow + 40));

  float rmax[8], rsum[8];
  v8f oacc[4];
#pragma unroll
  for (int r = 0; r < 8; ++r) { rmax[r] = -INFINITY; rsum[r] = 0.0f; }
#pragma unroll
  for (int t = 0; t < 4; ++t) oacc[t] = (v8f){};

  for (int m0 = 0; m0 < NPIX; m0 += 32) {
    // ---- scores: two 16x16 tiles (cols m0+lx and m0+16+lx) ----
    const __bf16* prow0 = pp + (size_t)(m0 + lx)      * C_CH + (hi ? 16 : 0);
    const __bf16* prow1 = pp + (size_t)(m0 + 16 + lx) * C_CH + (hi ? 16 : 0);
    if (m0 + 32 < NPIX)  // stream next phi tile toward L2
      __builtin_prefetch(pp + (size_t)(m0 + 32 + lx) * C_CH, 0, 1);

    v8f s0 = {}, s1 = {};
    s0 = wmma_bf16(a0, pack16(*(const v8bf*)(prow0 +  0), *(const v8bf*)(prow0 +  8)), s0);
    s0 = wmma_bf16(a1, pack16(*(const v8bf*)(prow0 + 32), *(const v8bf*)(prow0 + 40)), s0);
    s1 = wmma_bf16(a0, pack16(*(const v8bf*)(prow1 +  0), *(const v8bf*)(prow1 +  8)), s1);
    s1 = wmma_bf16(a1, pack16(*(const v8bf*)(prow1 + 32), *(const v8bf*)(prow1 + 40)), s1);

    // ---- online softmax per row (row = r + 8*hi, cols across 16 lanes) ----
#pragma unroll
    for (int r = 0; r < 8; ++r) {
      float m = fmaxf(s0[r], s1[r]);
#pragma unroll
      for (int d = 1; d < 16; d <<= 1) m = fmaxf(m, __shfl_xor(m, d, 32));
      const float nmax  = fmaxf(rmax[r], m);
      const float scale = __expf(rmax[r] - nmax);
      const float p0 = __expf(s0[r] - nmax);
      const float p1 = __expf(s1[r] - nmax);
      float psum = p0 + p1;
#pragma unroll
      for (int d = 1; d < 16; d <<= 1) psum += __shfl_xor(psum, d, 32);
      rsum[r] = rsum[r] * scale + psum;
      rmax[r] = nmax;
#pragma unroll
      for (int t = 0; t < 4; ++t) oacc[t][r] *= scale;
      // stage probabilities: [row][m-local]
      pstage[wid][r + hi * 8][lx]      = (__bf16)p0;
      pstage[wid][r + hi * 8][16 + lx] = (__bf16)p1;
    }
    asm volatile("s_wait_dscnt 0x0" ::: "memory");  // LDS RAW fence (wave-local)

    // ---- reload p as A fragment (16x32) ----
    const __bf16* pr = &pstage[wid][lx][hi ? 16 : 0];
    const v16bf pa = pack16(*(const v8bf*)(pr), *(const v8bf*)(pr + 8));

    // ---- out(16x64) += p(16x32) x g(32x16) per 16-col c tile ----
#pragma unroll
    for (int t = 0; t < 4; ++t) {
      const int c = t * 16 + lx;
      const __bf16* grow = gg + (size_t)c * NPIX + m0 + (hi ? 16 : 0);
      oacc[t] = wmma_bf16(pa, pack16(*(const v8bf*)(grow), *(const v8bf*)(grow + 8)),
                          oacc[t]);
    }
  }

  // ---- normalize and store y in (B, C, N) f32 ----
  float inv[8];
#pragma unroll
  for (int r = 0; r < 8; ++r) inv[r] = 1.0f / rsum[r];
#pragma unroll
  for (int t = 0; t < 4; ++t) {
    const int c = t * 16 + lx;
#pragma unroll
    for (int r = 0; r < 8; ++r) {
      const int n = n0 + r + hi * 8;
      yCN[((size_t)b * C_CH + c) * NPIX + n] = oacc[t][r] * inv[r];
    }
  }
}

// ---------------------------------------------------------------------------
extern "C" void kernel_launch(void* const* d_in, const int* in_sizes, int n_in,
                              void* d_out, int out_size, void* d_ws, size_t ws_size,
                              hipStream_t stream) {
  (void)in_sizes; (void)n_in; (void)out_size; (void)ws_size;
  const float* x    = (const float*)d_in[0];
  const float* g_w  = (const float*)d_in[1];
  const float* g_b  = (const float*)d_in[2];
  const float* th_w = (const float*)d_in[3];
  const float* th_b = (const float*)d_in[4];
  const float* ph_w = (const float*)d_in[5];
  const float* ph_b = (const float*)d_in[6];
  const float* W_w  = (const float*)d_in[7];
  const float* W_b  = (const float*)d_in[8];
  float* out = (float*)d_out;

  char* ws = (char*)d_ws;
  const size_t bfsz = (size_t)NBATCH * C_CH * NPIX * sizeof(__bf16);  // 1.6 MB
  __bf16* thetaT = (__bf16*)(ws);
  __bf16* phiT   = (__bf16*)(ws + bfsz);
  __bf16* gCN    = (__bf16*)(ws + 2 * bfsz);
  float*  yCN    = (float*)(ws + 3 * bfsz);

  const dim3 cgrid(NPIX / 128, C_CH / 16, NBATCH);  // (50, 4, 2)
  const dim3 cblk(256);

  conv3x3_wmma<<<cgrid, cblk, 0, stream>>>(x, th_w, th_b, nullptr, thetaT, nullptr,
                                           MODE_NC_BF16);
  conv3x3_wmma<<<cgrid, cblk, 0, stream>>>(x, ph_w, ph_b, nullptr, phiT, nullptr,
                                           MODE_NC_BF16);
  conv3x3_wmma<<<cgrid, cblk, 0, stream>>>(x, g_w, g_b, nullptr, gCN, nullptr,
                                           MODE_CN_BF16);

  nonlocal_attn_wmma<<<dim3(NBATCH * (NPIX / 16) / 4), dim3(128), 0, stream>>>(
      thetaT, phiT, gCN, yCN);

  conv3x3_wmma<<<cgrid, cblk, 0, stream>>>(yCN, W_w, W_b, x, nullptr, out,
                                           MODE_FINAL_F32);
}
